// AttentionMechanism_70437463655126
// MI455X (gfx1250) — matrix operand order repaired and checked by reading, not compiled, not run on hardware
//
#include <hip/hip_runtime.h>
#include <stdint.h>

#define B_ 32
#define T_ 2048
#define E_ 1024
#define D_ 1024
#define A_ 1024

// ext-vector types (trivially constructible -> safe in unions)
typedef __bf16 v16bf __attribute__((ext_vector_type(16)));
typedef float  v8f   __attribute__((ext_vector_type(8)));
typedef unsigned int u32x4 __attribute__((ext_vector_type(4)));

union ABfrag { v16bf v; u32x4 q[2]; };

__device__ __forceinline__ unsigned short f2bf(float f) {
  unsigned u = __float_as_uint(f);
  u += 0x7FFFu + ((u >> 16) & 1u);   // round-to-nearest-even
  return (unsigned short)(u >> 16);
}

// ---------------------------------------------------------------------------
// Kernel 1: W_enc [E,A] f32  ->  Wt [A,E] bf16  (transpose + convert via LDS)
// ---------------------------------------------------------------------------
__global__ void wenc_transpose_bf16(const float* __restrict__ W,
                                    unsigned short* __restrict__ Wt) {
  __shared__ float tile[32][33];
  const int a0 = blockIdx.x * 32, e0 = blockIdx.y * 32;
  const int tx = threadIdx.x, ty = threadIdx.y;
#pragma unroll
  for (int i = 0; i < 4; ++i) {
    int e = e0 + ty + i * 8;
    tile[ty + i * 8][tx] = W[(size_t)e * A_ + a0 + tx];
  }
  __syncthreads();
#pragma unroll
  for (int i = 0; i < 4; ++i) {
    int a = a0 + ty + i * 8;
    Wt[(size_t)a * E_ + e0 + tx] = f2bf(tile[tx][ty + i * 8]);
  }
}

// ---------------------------------------------------------------------------
// Kernel 2: base[b,a] = b_enc[a] + sum_d dec_out[b,d] * W_dec[d,a]
// ---------------------------------------------------------------------------
__global__ void base_kernel(const float* __restrict__ dec,
                            const float* __restrict__ Wd,
                            const float* __restrict__ be,
                            float* __restrict__ base) {
  const int b = blockIdx.y;
  const int a = blockIdx.x * 256 + threadIdx.x;
  const float* dp = dec + (size_t)b * D_;
  float s = be[a];
#pragma unroll 4
  for (int d = 0; d < D_; ++d) s += dp[d] * Wd[(size_t)d * A_ + a];
  base[(size_t)b * A_ + a] = s;
}

// ---------------------------------------------------------------------------
// Kernel 3: energy[b,t] = sum_a v[a] * tanh( enc[b,t,:]·W_enc[:,a] + base[b,a] )
// Block = 128 t-rows of one batch, 8 waves. Activation panel (128x1024 bf16,
// 258KB) lives in LDS for the whole block; wave w owns N-tiles 8w..8w+7 so the
// W panel is streamed from L2 exactly once per block. A fragments are
// prefetched 2 deep (4-slot rotating buffer), B tiles 1 K-step deep, so
// ds_load / global_load latency overlaps the WMMA pipe.
// ---------------------------------------------------------------------------
#define LSTR 1032   // padded LDS row stride in bf16 (2064 B -> bank-shifted rows)

__global__ __launch_bounds__(256, 1)
void energy_wmma(const float* __restrict__ enc,
                 const unsigned short* __restrict__ Wt,
                 const float* __restrict__ base,
                 const float* __restrict__ v,
                 float* __restrict__ energy) {
  extern __shared__ char smem[];
  float* e_lds = (float*)smem;                          // 128 floats (+pad)
  unsigned short* enc_lds = (unsigned short*)(smem + 2048);

  const int b    = blockIdx.x >> 4;          // T_/128 == 16 tiles per batch
  const int t0   = (blockIdx.x & 15) * 128;
  const int tid  = threadIdx.x;
  const int wave = tid >> 5;
  const int lane = tid & 31;
  const int lrow = lane & 15;
  const int khalf = lane >> 4;

  if (tid < 128) e_lds[tid] = 0.0f;

  // ---- stage 128 x 1024 activation rows as bf16 into LDS (coalesced) ----
  const float* encb = enc + ((size_t)b * T_ + t0) * E_;
#pragma unroll 4
  for (int row = 0; row < 128; ++row) {
    float4 x = ((const float4*)(encb + (size_t)row * E_))[tid];
    uint2 p;
    p.x = (unsigned)f2bf(x.x) | ((unsigned)f2bf(x.y) << 16);
    p.y = (unsigned)f2bf(x.z) | ((unsigned)f2bf(x.w) << 16);
    *((uint2*)(enc_lds + row * LSTR + tid * 4)) = p;
  }
  __syncthreads();

  // ---- WMMA main loop: wave owns 8 N-tiles, 8 M-subtiles each ----
  for (int ntl = 0; ntl < 8; ++ntl) {
    const int nt  = wave * 8 + ntl;
    const int col = nt * 16 + lrow;            // this lane's output column (a)
    const float vn = v[col];
    const float bb = base[(size_t)b * A_ + col];
    const unsigned short* Bp = Wt + (size_t)col * E_ + khalf * 8;

    auto loadB = [&](ABfrag& f, int k) {
      const u32x4* q = (const u32x4*)(Bp + k);
      f.q[0] = q[0];
      f.q[1] = *(const u32x4*)(Bp + k + 16);
    };
    auto loadA = [&](ABfrag& f, int m, int k) {
      const unsigned short* Ap =
          enc_lds + (m * 16 + lrow) * LSTR + k + khalf * 8;
      f.q[0] = *(const u32x4*)Ap;
      f.q[1] = *(const u32x4*)(Ap + 16);
    };

    // rotating fragment buffers: A prefetch distance 2, B distance 1 K-step.
    // All indices are compile-time constants after unrolling.
    ABfrag Bf[2], Af[4];
    loadB(Bf[0], 0);
    loadA(Af[0], 0, 0);
    loadA(Af[1], 1, 0);

    v8f acc[8] = {};
#pragma unroll 2
    for (int k0 = 0; k0 < E_; k0 += 32) {
      const int kb = (k0 >> 5) & 1;
      loadB(Bf[kb ^ 1], k0 + 32);     // prefetch next K-step's B tile
#pragma unroll
      for (int m = 0; m < 8; ++m) {
        const int pm = m + 2;          // prefetch A fragment 2 tiles ahead
        if (pm < 8) loadA(Af[pm & 3], pm, k0);
        else        loadA(Af[pm & 3], pm - 8, k0 + 32);
        acc[m] = __builtin_amdgcn_wmma_f32_16x16x32_bf16(
            false, Af[m & 3].v, false, Bf[kb].v, (short)0, acc[m], false, false);
      }
    }

    // ---- fused epilogue: +base, tanh, *v, reduce over the 16 N-lanes ----
#pragma unroll
    for (int m = 0; m < 8; ++m) {
#pragma unroll
      for (int r = 0; r < 8; ++r) {
        float s = vn * tanhf(acc[m][r] + bb);
        s += __shfl_xor(s, 1);
        s += __shfl_xor(s, 2);
        s += __shfl_xor(s, 4);
        s += __shfl_xor(s, 8);
        if (lrow == 0)
          atomicAdd(&e_lds[m * 16 + khalf * 8 + r], s);  // ds_add_f32
      }
    }
  }

  __syncthreads();
  if (tid < 128) energy[(size_t)b * T_ + t0 + tid] = e_lds[tid];
}

// ---------------------------------------------------------------------------
// Kernel 4: reference-exact mask (energy * 0/1 mask) + softmax over T
// ---------------------------------------------------------------------------
__global__ void softmax_mask(const float* __restrict__ energy,
                             const int* __restrict__ x_lens,
                             float* __restrict__ att) {
  __shared__ float red[8];
  const int b = blockIdx.x, tid = threadIdx.x;
  const int len = x_lens[b];
  float e[8];
  float mx = -1e30f;
#pragma unroll
  for (int i = 0; i < 8; ++i) {
    int t = i * 256 + tid;
    float raw = energy[(size_t)b * T_ + t];
    e[i] = (t < len) ? raw : 0.0f;             // energy * mask (NOT -inf)
    mx = fmaxf(mx, e[i]);
  }
  for (int m = 1; m < 32; m <<= 1) mx = fmaxf(mx, __shfl_xor(mx, m));
  if ((tid & 31) == 0) red[tid >> 5] = mx;
  __syncthreads();
  mx = red[0];
#pragma unroll
  for (int w = 1; w < 8; ++w) mx = fmaxf(mx, red[w]);
  __syncthreads();

  float sum = 0.0f;
#pragma unroll
  for (int i = 0; i < 8; ++i) { e[i] = expf(e[i] - mx); sum += e[i]; }
  for (int m = 1; m < 32; m <<= 1) sum += __shfl_xor(sum, m);
  if ((tid & 31) == 0) red[tid >> 5] = sum;
  __syncthreads();
  sum = 0.0f;
#pragma unroll
  for (int w = 0; w < 8; ++w) sum += red[w];
  const float inv = 1.0f / sum;
#pragma unroll
  for (int i = 0; i < 8; ++i)
    att[(size_t)b * T_ + i * 256 + tid] = e[i] * inv;
}

// ---------------------------------------------------------------------------
// Kernel 5: context[b,e] = sum_t att[b,t] * enc[b,t,e]   (bandwidth-bound)
// ---------------------------------------------------------------------------
__global__ void context_kernel(const float* __restrict__ enc,
                               const float* __restrict__ att,
                               float* __restrict__ ctx) {
  __shared__ float satt[T_];
  const int b = blockIdx.y, tid = threadIdx.x;
#pragma unroll
  for (int i = 0; i < 8; ++i)
    satt[i * 256 + tid] = att[(size_t)b * T_ + i * 256 + tid];
  __syncthreads();
  const int e = blockIdx.x * 256 + tid;
  const float* p = enc + (size_t)b * T_ * E_ + e;
  float acc = 0.0f;
#pragma unroll 8
  for (int t = 0; t < T_; ++t) acc += satt[t] * p[(size_t)t * E_];
  ctx[(size_t)b * E_ + e] = acc;
}

// ---------------------------------------------------------------------------
extern "C" void kernel_launch(void* const* d_in, const int* in_sizes, int n_in,
                              void* d_out, int out_size, void* d_ws, size_t ws_size,
                              hipStream_t stream) {
  (void)in_sizes; (void)n_in; (void)out_size; (void)ws_size;
  const float* enc    = (const float*)d_in[0];
  const int*   x_lens = (const int*)d_in[1];
  const float* dec    = (const float*)d_in[2];
  /* d_in[3] = att_weights_step: unused by the reference */
  const float* Wenc   = (const float*)d_in[4];
  const float* benc   = (const float*)d_in[5];
  const float* Wdec   = (const float*)d_in[6];
  const float* v      = (const float*)d_in[7];

  char* ws = (char*)d_ws;
  unsigned short* Wt  = (unsigned short*)ws;                         // 2 MB bf16
  float* base         = (float*)(ws + 2u * 1024u * 1024u);           // 128 KB
  float* energy       = (float*)(ws + 2u * 1024u * 1024u + 131072u); // 256 KB

  float* ctx_out = (float*)d_out;                    // context_vec [B,1,E]
  float* att_out = (float*)d_out + (size_t)B_ * E_;  // att [B,T]

  wenc_transpose_bf16<<<dim3(A_ / 32, E_ / 32), dim3(32, 8), 0, stream>>>(Wenc, Wt);
  base_kernel<<<dim3(A_ / 256, B_), 256, 0, stream>>>(dec, Wdec, benc, base);

  const size_t lds_bytes = 2048 + 128 * LSTR * 2;   // 266,240 B <= 320 KB/WGP
  energy_wmma<<<dim3(B_ * T_ / 128), 256, lds_bytes, stream>>>(enc, Wt, base, v, energy);

  softmax_mask<<<dim3(B_), 256, 0, stream>>>(energy, x_lens, att_out);
  context_kernel<<<dim3(E_ / 256, B_), 256, 0, stream>>>(enc, att_out, ctx_out);
}
